// TransformerBlock_39754217292164
// MI455X (gfx1250) — matrix operand order repaired and checked by reading, not compiled, not run on hardware
//
#include <hip/hip_runtime.h>
#include <hip/hip_bf16.h>
#include <math.h>

typedef __attribute__((ext_vector_type(16))) _Float16 v16h;
typedef __attribute__((ext_vector_type(8)))  _Float16 v8h;
typedef __attribute__((ext_vector_type(8)))  float    v8f;

#if __has_builtin(__builtin_amdgcn_tensor_load_to_lds) && \
    __has_builtin(__builtin_amdgcn_s_wait_tensorcnt)
#define USE_TDM 1
#else
#define USE_TDM 0
#endif

typedef unsigned int u32x4 __attribute__((ext_vector_type(4)));
typedef int          i32x8 __attribute__((ext_vector_type(8)));
typedef int          i32x4 __attribute__((ext_vector_type(4)));

// ---------------------------------------------------------------------------
// WMMA helpers (CDNA5 wave32 layouts, 16x16x32 f16 -> f32)
// ---------------------------------------------------------------------------
__device__ __forceinline__ v8f wmma16(v16h a, v16h b, v8f c) {
  return __builtin_amdgcn_wmma_f32_16x16x32_f16(false, a, false, b, (short)0, c,
                                                false, false);
}

// A fragment, generic strided/zero-padded form.
template <typename T>
__device__ __forceinline__ v16h ld_a(const T* s, int rs, int ks, int kmax,
                                     float sc) {
  int l = threadIdx.x & 31, r = l & 15, hf = l >> 4;
  const T* row = s + (size_t)r * rs;
  v16h a;
#pragma unroll
  for (int i = 0; i < 8; ++i) {
    int k0 = hf * 16 + i, k1 = hf * 16 + 8 + i;
    a[i]     = (k0 < kmax) ? (_Float16)((float)row[(size_t)k0 * ks] * sc)
                           : (_Float16)0.f;
    a[8 + i] = (k1 < kmax) ? (_Float16)((float)row[(size_t)k1 * ks] * sc)
                           : (_Float16)0.f;
  }
  return a;
}

// A fragment, contiguous-K fast path: two 16B vector loads per lane.
// Requires rs (in halfs) multiple of 8 and 16B-aligned base.
__device__ __forceinline__ v16h ld_a_c(const _Float16* s, int rs) {
  int l = threadIdx.x & 31, r = l & 15, hf = l >> 4;
  const _Float16* row = s + (size_t)r * rs + hf * 16;
  v8h lo = *(const v8h*)row;
  v8h hi = *(const v8h*)(row + 8);
  v16h a;
#pragma unroll
  for (int i = 0; i < 8; ++i) { a[i] = lo[i]; a[8 + i] = hi[i]; }
  return a;
}

// B fragment: 32x16 (KxN). lane<16: col=lane, K=0..15 ; lane>=16: col=lane-16,
// K=16..31.  ks = K stride, ns = col stride, kmax = valid K.
template <typename T>
__device__ __forceinline__ v16h ld_b(const T* s, int ks, int ns, int kmax) {
  int l = threadIdx.x & 31, r = l & 15, hf = l >> 4;
  v16h b;
#pragma unroll
  for (int j = 0; j < 16; ++j) {
    int k = hf * 16 + j;
    b[j] = (k < kmax) ? (_Float16)(float)s[(size_t)k * ks + (size_t)r * ns]
                      : (_Float16)0.f;
  }
  return b;
}

// B fragment from pre-swizzled storage: tile stored as [lane][16 halfs],
// each lane reads its 32 contiguous bytes (two 16B vector loads).
__device__ __forceinline__ v16h ld_b_sw(const _Float16* tile) {
  int l = threadIdx.x & 31;
  const _Float16* p = tile + l * 16;
  v8h lo = *(const v8h*)p;
  v8h hi = *(const v8h*)(p + 8);
  v16h b;
#pragma unroll
  for (int i = 0; i < 8; ++i) { b[i] = lo[i]; b[8 + i] = hi[i]; }
  return b;
}

// C/D fragment store: VGPR v -> (M = v + 8*half, N = lane&15)
__device__ __forceinline__ void st_c(_Float16* d, int ld, v8f c) {
  int l = threadIdx.x & 31, r = l & 15, hf = l >> 4;
#pragma unroll
  for (int v = 0; v < 8; ++v) d[(size_t)(v + 8 * hf) * ld + r] = (_Float16)c[v];
}

__device__ __forceinline__ float gelu_exact(float x) {
  return 0.5f * x * (1.f + erff(x * 0.70710678118654752f));
}

// Swin shifted-window mask, computed analytically (H=W=256, ws=8, shift=4)
__device__ __forceinline__ int regid(int p) {
  return p < 248 ? 0 : (p < 252 ? 1 : 2);
}
__device__ __forceinline__ float maskval(int wy, int wx, int tm, int tn) {
  int im = regid(wy * 8 + (tm >> 3)) * 3 + regid(wx * 8 + (tm & 7));
  int jn = regid(wy * 8 + (tn >> 3)) * 3 + regid(wx * 8 + (tn & 7));
  return (im == jn) ? 0.f : -100.f;
}

// Swizzle a (K x N) row-major f32 weight into B-fragment layout:
// dst[(((nt*nkt)+kt)*32 + lane)*16 + j] = W[kt*32 + (lane>>4)*16 + j][nt*16+(lane&15)]
__device__ __forceinline__ void swizzle_b(const float* w, _Float16* dst, int K,
                                          int N, int tid, int st) {
  int nkt = K >> 5;
  int total = K * N;
  for (int i = tid; i < total; i += st) {
    int j = i & 15;
    int l = (i >> 4) & 31;
    int t = i >> 9;  // nt*nkt + kt
    int kt = t % nkt, nt = t / nkt;
    int k = kt * 32 + (l >> 4) * 16 + j;
    int n = nt * 16 + (l & 15);
    dst[i] = (_Float16)w[k * N + n];
  }
}

#if USE_TDM
// Tensor Data Mover: stage a (krows x 32) f16 panel (global row stride 65536
// elements) into LDS.  One TDM op per block, issued by one wave.
// 6-arg builtin: (g0, g1, g2, g3, g4, cpol)
__device__ __forceinline__ void tdm_load_panel(const _Float16* gsrc,
                                               void* lds_dst, unsigned krows) {
  unsigned long long ga = (unsigned long long)(uintptr_t)gsrc;
  unsigned lo = (unsigned)(uintptr_t)lds_dst;  // low 32 bits = LDS offset
  u32x4 g0;
  g0[0] = 1u;                                   // count=1 user descriptor
  g0[1] = lo;                                   // lds_addr
  g0[2] = (unsigned)(ga & 0xffffffffu);         // global_addr[31:0]
  g0[3] = (unsigned)((ga >> 32) & 0x1ffffffu)   // global_addr[56:32]
          | (2u << 30);                         // type = 2 (image)
  i32x8 g1;
  g1[0] = 1 << 16;                              // data_size = 2B, no multicast
  g1[1] = 0;                                    // tensor_dim0 lo16 (=0)
  g1[2] = 1 | (int)(krows << 16);               // dim0 hi16 (=1 -> 65536),
                                                // tensor_dim1 lo16 = krows
  g1[3] = (32 << 16);                           // dim1 hi16=0, tile_dim0 = 32
  g1[4] = (int)krows;                           // tile_dim1 = krows, tile_dim2=0
  g1[5] = 65536;                                // tensor_dim0_stride lo32
  g1[6] = 0;                                    // stride0 hi, stride1 lo
  g1[7] = 0;                                    // stride1 hi
  i32x4 z4 = {0, 0, 0, 0};
  i32x8 z8 = {0, 0, 0, 0, 0, 0, 0, 0};
  __builtin_amdgcn_tensor_load_to_lds(g0, g1, z4, z4, z8, 0);
}
#endif

// Stage B panel (krows x 32, row stride 65536) to LDS, all waves synced after.
__device__ __forceinline__ void stage_panel(const _Float16* gsrc,
                                            _Float16* lds, int krows) {
#if USE_TDM
  if (threadIdx.x < 32) {
    tdm_load_panel(gsrc, lds, (unsigned)krows);
    __builtin_amdgcn_s_wait_tensorcnt(0);
  }
#else
  for (int i = threadIdx.x; i < krows * 32; i += 256)
    lds[i] = gsrc[(size_t)(i >> 5) * 65536 + (i & 31)];
#endif
  __syncthreads();
}

// ---------------------------------------------------------------------------
// K0: weight conversion (B-operands pre-swizzled) + sine positional table
// ---------------------------------------------------------------------------
__global__ void k0_prep(const float* wqkv, const float* w11, const float* w12,
                        const float* w2, const float* c11, const float* ffin,
                        const float* ffout, _Float16* wqkvsw, _Float16* w11sw,
                        _Float16* w12sw, _Float16* w2sw, _Float16* c11h,
                        _Float16* ffinh, _Float16* ffouth, float* pos) {
  int tid = blockIdx.x * blockDim.x + threadIdx.x;
  int st = gridDim.x * blockDim.x;
  // B operands -> fragment-swizzled f16
  swizzle_b(wqkv, wqkvsw, 64, 192, tid, st);
  for (int e = 0; e < 3; ++e) {
    swizzle_b(w11 + e * 12288, w11sw + e * 12288, 192, 64, tid, st);
    swizzle_b(w12 + e * 12288, w12sw + e * 12288, 192, 64, tid, st);
    swizzle_b(w2 + e * 4096, w2sw + e * 4096, 64, 64, tid, st);
  }
  // A operands -> row-major f16
  for (int i = tid; i < 36864; i += st)  c11h[i]   = (_Float16)c11[i];
  for (int i = tid; i < 147456; i += st) ffinh[i]  = (_Float16)ffin[i];
  for (int i = tid; i < 73728; i += st)  ffouth[i] = (_Float16)ffout[i];
  for (int i = tid; i < 4096; i += st) {
    int n = i >> 6, d = i & 63;
    int ty = n >> 3, tx = n & 7;
    float coord = (d < 32) ? (float)(ty + 1) : (float)(tx + 1);
    int dd = d & 31, j = dd >> 1;
    float dim = powf(10000.f, (float)j / 16.f);
    float v = coord / (8.f + 1e-6f) * 6.283185307179586f;
    float arg = v / dim;
    pos[i] = (dd & 1) ? cosf(arg) : sinf(arg);
  }
}

// ---------------------------------------------------------------------------
// K1: window partition (roll -4 folded) + LN1 + pos + QKV GEMM (64x64 @ 64x192)
// grid = 3*1024, block = 128 (4 waves)
// ---------------------------------------------------------------------------
__global__ void k1_ln_qkv(const float* __restrict__ x, const float* ln1w,
                          const float* ln1b,
                          const _Float16* __restrict__ wqkvsw,
                          const float* __restrict__ pos,
                          _Float16* __restrict__ qkvbuf) {
  __shared__ float    xr[64 * 64];
  __shared__ _Float16 xln[64 * 80];
  int bid = blockIdx.x;
  int e = bid >> 10, win = bid & 1023;
  int wy = win >> 5, wx = win & 31;
  int tid = threadIdx.x;
  for (int i = tid; i < 4096; i += 128) {
    int c = i >> 6, n = i & 63;
    int hs = (wy * 8 + (n >> 3) + 4) & 255;
    int ws = (wx * 8 + (n & 7) + 4) & 255;
    xr[n * 64 + c] = x[(size_t)((e * 64 + c) * 256 + hs) * 256 + ws];
  }
  __syncthreads();
  if (tid < 64) {
    float m = 0.f;
    for (int c = 0; c < 64; ++c) m += xr[tid * 64 + c];
    m *= (1.f / 64.f);
    float v = 0.f;
    for (int c = 0; c < 64; ++c) { float d = xr[tid * 64 + c] - m; v += d * d; }
    v *= (1.f / 64.f);
    float rs = rsqrtf(v + 1e-5f);
    for (int c = 0; c < 64; ++c)
      xln[tid * 80 + c] = (_Float16)((xr[tid * 64 + c] - m) * rs * ln1w[c] +
                                     ln1b[c] + pos[tid * 64 + c]);
  }
  __syncthreads();
  int wave = tid >> 5;
  _Float16* out = qkvbuf + (size_t)bid * 64 * 192;
  for (int t = wave; t < 48; t += 4) {
    int mt = t & 3, nt = t >> 2;  // 4 x 12 tiles
    v8f acc = {};
#pragma unroll
    for (int kt = 0; kt < 2; ++kt) {
      v16h a = ld_a_c(&xln[mt * 16 * 80 + kt * 32], 80);
      v16h b = ld_b_sw(wqkvsw + (size_t)(nt * 2 + kt) * 512);
      acc = wmma16(a, b, acc);
    }
    st_c(out + (mt * 16) * 192 + nt * 16, 192, acc);
  }
}

// ---------------------------------------------------------------------------
// K2: window attention, one wave per head.  grid = 6*1024, block = 128.
// combos: 0:(q1,k0) 1:(q1,k2) 2:(q0,k1) 3:(q0,k2) 4:(q2,k1) 5:(q2,k0)
// head split is head-fast: feature f = c*4 + h ; merged out g = h*16 + c
// ---------------------------------------------------------------------------
__global__ void k2_attn(const _Float16* __restrict__ qkvbuf,
                        _Float16* __restrict__ attbuf) {
  __shared__ float sld[4 * 64 * 64];  // 64KB: per-head f32 score matrix
  static const int QE[6] = {1, 1, 0, 0, 2, 2};
  static const int KE[6] = {0, 2, 1, 2, 1, 0};
  int bid = blockIdx.x;
  int combo = bid >> 10, win = bid & 1023;
  int wy = win >> 5, wx = win & 31;
  int h = threadIdx.x >> 5;
  int lane = threadIdx.x & 31, r = lane & 15, hf = lane >> 4;
  const _Float16* qb = qkvbuf + (size_t)(QE[combo] * 1024 + win) * 64 * 192 + h;
  const _Float16* kb =
      qkvbuf + (size_t)(KE[combo] * 1024 + win) * 64 * 192 + 64 + h;
  const _Float16* vb = kb + 64;
  float* S = &sld[h * 64 * 64];

  // S = (q * fnum^-0.5) @ k^T + mask   (K=16 zero-padded to 32)
  for (int mt = 0; mt < 4; ++mt) {
    v16h a = ld_a(qb + mt * 16 * 192, 192, 4, 16, 0.125f);
    for (int nt = 0; nt < 4; ++nt) {
      v16h b = ld_b(kb + nt * 16 * 192, 4, 192, 16);
      v8f z = {};
      v8f s = wmma16(a, b, z);
#pragma unroll
      for (int v = 0; v < 8; ++v) {
        int tm = mt * 16 + v + 8 * hf, tn = nt * 16 + r;
        S[tm * 64 + tn] = s[v] + maskval(wy, wx, tm, tn);
      }
    }
  }
  __syncthreads();
  // softmax: each lane owns rows lane, lane+32
  for (int rr = lane; rr < 64; rr += 32) {
    float mx = -1e30f;
    for (int cc = 0; cc < 64; ++cc) mx = fmaxf(mx, S[rr * 64 + cc]);
    float sum = 0.f;
    for (int cc = 0; cc < 64; ++cc) {
      float ev = __expf(S[rr * 64 + cc] - mx);
      S[rr * 64 + cc] = ev;
      sum += ev;
    }
    float inv = 1.f / sum;
    for (int cc = 0; cc < 64; ++cc) S[rr * 64 + cc] *= inv;
  }
  __syncthreads();
  // out = P @ v  (64x64 @ 64x16), merged feature g = h*16 + c
  _Float16* ob = attbuf + (size_t)bid * 64 * 64 + h * 16;
  for (int mt = 0; mt < 4; ++mt) {
    v8f acc = {};
#pragma unroll
    for (int kt = 0; kt < 2; ++kt) {
      v16h a = ld_a(&S[mt * 16 * 64 + kt * 32], 64, 1, 32, 1.f);
      v16h b = ld_b(vb + kt * 32 * 192, 192, 4, 32);
      acc = wmma16(a, b, acc);
    }
    st_c(ob + (mt * 16) * 64, 64, acc);
  }
}

// ---------------------------------------------------------------------------
// K3: concat(att_a, att_b, merged q) -> GEGLU -> 64x64 ; scatter to channel-
// major (192 x 65536) f16 image in shifted frame.  grid = 3*1024, block = 128.
// ---------------------------------------------------------------------------
__global__ void k3_geglu(const _Float16* __restrict__ qkvbuf,
                         const _Float16* __restrict__ attbuf,
                         const _Float16* __restrict__ w11sw, const float* b11,
                         const _Float16* __restrict__ w12sw, const float* b12,
                         const _Float16* __restrict__ w2sw, const float* b2,
                         _Float16* __restrict__ xc) {
  __shared__ _Float16 X[64 * 208];
  __shared__ _Float16 P[64 * 80];
  static const int CA[3] = {2, 0, 4}, CB[3] = {3, 1, 5};
  int bid = blockIdx.x;
  int e = bid >> 10, win = bid & 1023;
  int wy = win >> 5, wx = win & 31;
  int tid = threadIdx.x;
  const _Float16* a0 = attbuf + (size_t)(CA[e] * 1024 + win) * 64 * 64;
  const _Float16* a1 = attbuf + (size_t)(CB[e] * 1024 + win) * 64 * 64;
  const _Float16* qb = qkvbuf + (size_t)(e * 1024 + win) * 64 * 192;
  // fill concat X: 64 rows x 192 feats, in 8-half chunks (24 per row)
  for (int cch = tid; cch < 64 * 24; cch += 128) {
    int n = cch / 24, f0 = (cch % 24) * 8;
    v8h val;
    if (f0 < 64) {
      val = *(const v8h*)(a0 + n * 64 + f0);
    } else if (f0 < 128) {
      val = *(const v8h*)(a1 + n * 64 + (f0 - 64));
    } else {
#pragma unroll
      for (int j = 0; j < 8; ++j) {
        int g = f0 - 128 + j;
        val[j] = qb[n * 192 + (g & 15) * 4 + (g >> 4)];
      }
    }
    *(v8h*)(&X[n * 208 + f0]) = val;
  }
  __syncthreads();
  int wave = tid >> 5, lane = tid & 31, r = lane & 15, hf = lane >> 4;
  const _Float16* W11 = w11sw + e * 12288;
  const _Float16* W12 = w12sw + e * 12288;
  for (int t = wave; t < 16; t += 4) {
    int mt = t & 3, nt = t >> 2;
    v8f g1 = {}, g2 = {};
#pragma unroll
    for (int kt = 0; kt < 6; ++kt) {
      v16h a = ld_a_c(&X[mt * 16 * 208 + kt * 32], 208);
      g1 = wmma16(a, ld_b_sw(W11 + (size_t)(nt * 6 + kt) * 512), g1);
      g2 = wmma16(a, ld_b_sw(W12 + (size_t)(nt * 6 + kt) * 512), g2);
    }
    int o = nt * 16 + r;
    float bb1 = b11[e * 64 + o], bb2 = b12[e * 64 + o];
#pragma unroll
    for (int v = 0; v < 8; ++v) {
      float x1 = g1[v] + bb1, x2 = g2[v] + bb2;
      P[(mt * 16 + v + 8 * hf) * 80 + o] = (_Float16)(gelu_exact(x1) * x2);
    }
  }
  __syncthreads();
  const _Float16* W2 = w2sw + e * 4096;
  for (int t = wave; t < 16; t += 4) {
    int mt = t & 3, nt = t >> 2;
    v8f acc = {};
#pragma unroll
    for (int kt = 0; kt < 2; ++kt) {
      v16h a = ld_a_c(&P[mt * 16 * 80 + kt * 32], 80);
      acc = wmma16(a, ld_b_sw(W2 + (size_t)(nt * 2 + kt) * 512), acc);
    }
    float bb = b2[e * 64 + nt * 16 + r];
#pragma unroll
    for (int v = 0; v < 8; ++v) {
      int n = mt * 16 + v + 8 * hf;
      int hs = wy * 8 + (n >> 3), ws = wx * 8 + (n & 7);
      xc[(size_t)(e * 64 + nt * 16 + r) * 65536 + hs * 256 + ws] =
          (_Float16)(acc[v] + bb);
    }
  }
}

// ---------------------------------------------------------------------------
// K4: 1x1 conv (192x192) over pixels + roll(+4) + residual -> xres (f32)
// block: 256 thr, TDM-staged 192x32 B panel, 24 tiles.  grid = 2048.
// ---------------------------------------------------------------------------
__global__ void k4_conv11(const _Float16* __restrict__ c11h,
                          const _Float16* __restrict__ xc,
                          const float* __restrict__ x,
                          float* __restrict__ xres) {
  __shared__ _Float16 Bp[192 * 32];
  int p0 = blockIdx.x * 32;
  stage_panel(xc + p0, Bp, 192);
  int wave = threadIdx.x >> 5, lane = threadIdx.x & 31, r = lane & 15,
      hf = lane >> 4;
  for (int t = wave; t < 24; t += 8) {
    int mt = t % 12, ct2 = t / 12;
    v8f acc = {};
#pragma unroll
    for (int kt = 0; kt < 6; ++kt) {
      v16h a = ld_a_c(c11h + (mt * 16) * 192 + kt * 32, 192);
      v16h b = ld_b(&Bp[kt * 32 * 32 + ct2 * 16], 32, 1, 32);
      acc = wmma16(a, b, acc);
    }
#pragma unroll
    for (int v = 0; v < 8; ++v) {
      int o = mt * 16 + v + 8 * hf;
      int p = p0 + ct2 * 16 + r;
      int hs = p >> 8, ws = p & 255;
      int pf = (((hs + 4) & 255) << 8) | ((ws + 4) & 255);
      size_t oi = (size_t)o * 65536 + pf;
      xres[oi] = acc[v] + x[oi];
    }
  }
}

// ---------------------------------------------------------------------------
// K5: LN2 over 192 channels per pixel -> channel-major f16
// ---------------------------------------------------------------------------
__global__ void k5_ln2(const float* __restrict__ xres, const float* ln2w,
                       const float* ln2b, _Float16* __restrict__ xln) {
  int p = blockIdx.x * blockDim.x + threadIdx.x;  // 65536 pixels
  float m = 0.f;
  for (int c = 0; c < 192; ++c) m += xres[(size_t)c * 65536 + p];
  m *= (1.f / 192.f);
  float v = 0.f;
  for (int c = 0; c < 192; ++c) {
    float d = xres[(size_t)c * 65536 + p] - m;
    v += d * d;
  }
  v *= (1.f / 192.f);
  float rs = rsqrtf(v + 1e-5f);
  for (int c = 0; c < 192; ++c)
    xln[(size_t)c * 65536 + p] =
        (_Float16)((xres[(size_t)c * 65536 + p] - m) * rs * ln2w[c] + ln2b[c]);
}

// ---------------------------------------------------------------------------
// K6: ff_in GEMM (768x192 @ 192x65536).  TDM-staged panel, 96 tiles/block.
// grid = 2048, block = 256.
// ---------------------------------------------------------------------------
__global__ void k6_ffin(const _Float16* __restrict__ ffinh,
                        const _Float16* __restrict__ xln,
                        _Float16* __restrict__ ffh) {
  __shared__ _Float16 Bp[192 * 32];
  int p0 = blockIdx.x * 32;
  stage_panel(xln + p0, Bp, 192);
  int wave = threadIdx.x >> 5, lane = threadIdx.x & 31, r = lane & 15,
      hf = lane >> 4;
  for (int t = wave; t < 96; t += 8) {
    int mt = t % 48, ct2 = t / 48;
    v8f acc = {};
#pragma unroll
    for (int kt = 0; kt < 6; ++kt) {
      v16h a = ld_a_c(ffinh + (mt * 16) * 192 + kt * 32, 192);
      v16h b = ld_b(&Bp[kt * 32 * 32 + ct2 * 16], 32, 1, 32);
      acc = wmma16(a, b, acc);
    }
#pragma unroll
    for (int v = 0; v < 8; ++v)
      ffh[(size_t)(mt * 16 + v + 8 * hf) * 65536 + p0 + ct2 * 16 + r] =
          (_Float16)acc[v];
  }
}

// ---------------------------------------------------------------------------
// K7: depthwise 3x3 (768 ch, pad 1) + split + exact-gelu gate -> 384 ch f16
// ---------------------------------------------------------------------------
__global__ void k7_dwgate(const _Float16* __restrict__ ffh,
                          const float* __restrict__ dww,
                          _Float16* __restrict__ g) {
  int i = blockIdx.x * blockDim.x + threadIdx.x;
  if (i >= 384 * 65536) return;
  int c = i >> 16, p = i & 65535;
  int y = p >> 8, xw = p & 255;
  float s1 = 0.f, s2 = 0.f;
#pragma unroll
  for (int dy = 0; dy < 3; ++dy) {
    int yy = y + dy - 1;
    if ((unsigned)yy > 255u) continue;
#pragma unroll
    for (int dx = 0; dx < 3; ++dx) {
      int xx = xw + dx - 1;
      if ((unsigned)xx > 255u) continue;
      int pp = (yy << 8) | xx;
      s1 += dww[(c * 3 + dy) * 3 + dx] * (float)ffh[(size_t)c * 65536 + pp];
      s2 += dww[((c + 384) * 3 + dy) * 3 + dx] *
            (float)ffh[(size_t)(c + 384) * 65536 + pp];
    }
  }
  g[i] = (_Float16)(gelu_exact(s1) * s2);
}

// ---------------------------------------------------------------------------
// K8: ff_out GEMM (192x384 @ 384x65536) + residual -> d_out (f32)
// TDM-staged 384x32 panel, 24 tiles/block.  grid = 2048, block = 256.
// ---------------------------------------------------------------------------
__global__ void k8_ffout(const _Float16* __restrict__ ffouth,
                         const _Float16* __restrict__ g,
                         const float* __restrict__ xres,
                         float* __restrict__ out) {
  __shared__ _Float16 Bp[384 * 32];
  int p0 = blockIdx.x * 32;
  stage_panel(g + p0, Bp, 384);
  int wave = threadIdx.x >> 5, lane = threadIdx.x & 31, r = lane & 15,
      hf = lane >> 4;
  for (int t = wave; t < 24; t += 8) {
    int mt = t % 12, ct2 = t / 12;
    v8f acc = {};
#pragma unroll
    for (int kt = 0; kt < 12; ++kt) {
      v16h a = ld_a_c(ffouth + (mt * 16) * 384 + kt * 32, 384);
      v16h b = ld_b(&Bp[kt * 32 * 32 + ct2 * 16], 32, 1, 32);
      acc = wmma16(a, b, acc);
    }
#pragma unroll
    for (int v = 0; v < 8; ++v) {
      size_t oi =
          (size_t)(mt * 16 + v + 8 * hf) * 65536 + p0 + ct2 * 16 + r;
      out[oi] = xres[oi] + acc[v];
    }
  }
}

// ---------------------------------------------------------------------------
extern "C" void kernel_launch(void* const* d_in, const int* in_sizes, int n_in,
                              void* d_out, int out_size, void* d_ws,
                              size_t ws_size, hipStream_t stream) {
  const float* x    = (const float*)d_in[0];
  const float* ln1w = (const float*)d_in[1];
  const float* ln1b = (const float*)d_in[2];
  const float* ln2w = (const float*)d_in[3];
  const float* ln2b = (const float*)d_in[4];
  const float* wqkv = (const float*)d_in[5];
  const float* w11  = (const float*)d_in[6];
  const float* b11  = (const float*)d_in[7];
  const float* w12  = (const float*)d_in[8];
  const float* b12  = (const float*)d_in[9];
  const float* w2   = (const float*)d_in[10];
  const float* b2   = (const float*)d_in[11];
  const float* c11  = (const float*)d_in[12];
  const float* ffin = (const float*)d_in[13];
  const float* dww  = (const float*)d_in[14];
  const float* ffout = (const float*)d_in[15];
  float* out = (float*)d_out;

  char* ws = (char*)d_ws;
  size_t off = 0;
  auto take = [&](size_t bytes) -> char* {
    char* p = ws + off;
    off = (off + bytes + 255) & ~(size_t)255;
    return p;
  };
  _Float16* wqkvsw = (_Float16*)take(12288 * 2);
  _Float16* w11sw  = (_Float16*)take(36864 * 2);
  _Float16* w12sw  = (_Float16*)take(36864 * 2);
  _Float16* w2sw   = (_Float16*)take(12288 * 2);
  _Float16* c11h   = (_Float16*)take(36864 * 2);
  _Float16* ffinh  = (_Float16*)take(147456 * 2);
  _Float16* ffouth = (_Float16*)take(73728 * 2);
  float*    pos    = (float*)take(4096 * 4);
  // lifetime-aliased regions
  char* regionA = take(100663296);  // qkv f16 (75.5MB) then ffh f16 (100.7MB)
  char* regionB = take(50331648);   // att f16 then gate f16
  char* regionC = take(25165824);   // xc f16 then xln f16
  float* xres   = (float*)take(50331648);

  _Float16* qkvbuf = (_Float16*)regionA;
  _Float16* ffh    = (_Float16*)regionA;
  _Float16* attbuf = (_Float16*)regionB;
  _Float16* gbuf   = (_Float16*)regionB;
  _Float16* xc     = (_Float16*)regionC;
  _Float16* xlnb   = (_Float16*)regionC;

  k0_prep<<<512, 256, 0, stream>>>(wqkv, w11, w12, w2, c11, ffin, ffout,
                                   wqkvsw, w11sw, w12sw, w2sw, c11h, ffinh,
                                   ffouth, pos);
  k1_ln_qkv<<<3072, 128, 0, stream>>>(x, ln1w, ln1b, wqkvsw, pos, qkvbuf);
  k2_attn<<<6144, 128, 0, stream>>>(qkvbuf, attbuf);
  k3_geglu<<<3072, 128, 0, stream>>>(qkvbuf, attbuf, w11sw, b11, w12sw, b12,
                                     w2sw, b2, xc);
  k4_conv11<<<2048, 256, 0, stream>>>(c11h, xc, x, xres);
  k5_ln2<<<256, 256, 0, stream>>>(xres, ln2w, ln2b, xlnb);
  k6_ffin<<<2048, 256, 0, stream>>>(ffinh, xlnb, ffh);
  k7_dwgate<<<98304, 256, 0, stream>>>(ffh, dww, gbuf);
  k8_ffout<<<2048, 256, 0, stream>>>(ffouth, gbuf, xres, out);
}
